// EncoderGraphConv_80015240725034
// MI455X (gfx1250) — matrix operand order repaired and checked by reading, not compiled, not run on hardware
//
#include <hip/hip_runtime.h>
#include <hip/hip_bf16.h>

// Problem dimensions (fixed by the reference).
#define NN 65536
#define FF 512
#define HH 512
#define EE 524288

typedef __attribute__((ext_vector_type(16))) __bf16 v16bf;
typedef __attribute__((ext_vector_type(8)))  __bf16 v8bf;
typedef __attribute__((ext_vector_type(8)))  float  v8f;

// ---------------------------------------------------------------------------
// Zero-fill (graph-capture safe)
// ---------------------------------------------------------------------------
__global__ void fill_zero_f32(float* __restrict__ p, size_t n) {
  size_t i = (size_t)blockIdx.x * blockDim.x + threadIdx.x;
  size_t stride = (size_t)gridDim.x * blockDim.x;
  for (; i < n; i += stride) p[i] = 0.0f;
}

// ---------------------------------------------------------------------------
// f32 -> bf16 conversion, 8 elements per thread (n must be a multiple of 8)
// ---------------------------------------------------------------------------
__global__ void __launch_bounds__(256)
cvt_f32_bf16(const float* __restrict__ src, __bf16* __restrict__ dst, size_t n) {
  size_t i = ((size_t)blockIdx.x * blockDim.x + threadIdx.x) * 8;
  size_t stride = (size_t)gridDim.x * blockDim.x * 8;
  for (; i < n; i += stride) {
    const float4 a = *(const float4*)(src + i);
    const float4 b = *(const float4*)(src + i + 4);
    v8bf o;
    o[0] = (__bf16)a.x; o[1] = (__bf16)a.y; o[2] = (__bf16)a.z; o[3] = (__bf16)a.w;
    o[4] = (__bf16)b.x; o[5] = (__bf16)b.y; o[6] = (__bf16)b.z; o[7] = (__bf16)b.w;
    *(v8bf*)(dst + i) = o;
  }
}

// ---------------------------------------------------------------------------
// Edge aggregation: agg[dst] += h[src]  (bf16 source rows, f32 atomic accum
// into an L2-resident 128MB buffer). One thread per (edge, 8-elem chunk).
// ---------------------------------------------------------------------------
__global__ void __launch_bounds__(256)
aggregate_bf16(const __bf16* __restrict__ h,
               const int* __restrict__ src,
               const int* __restrict__ dst,
               float* __restrict__ agg) {
  const size_t idx = (size_t)blockIdx.x * blockDim.x + threadIdx.x;
  const size_t total = (size_t)EE * (FF / 8);
  if (idx >= total) return;
  const int e = (int)(idx >> 6);   // / 64 chunks per row
  const int c = (int)(idx & 63);
  const int s = src[e];
  const int d = dst[e];
  const v8bf v = *(const v8bf*)(h + (size_t)s * FF + (size_t)c * 8);
  float* o = agg + (size_t)d * FF + (size_t)c * 8;
#pragma unroll
  for (int i = 0; i < 8; ++i) atomicAdd(o + i, (float)v[i]);
}

// ---------------------------------------------------------------------------
// WMMA bf16 fragment loads (pre-converted bf16 operands, K stride = 512).
//
// A (16x32 bf16): lane L holds row m = L&15; elements 0..15 are
// K = kb+16h..+15 (h = L>>4), matching the B fragment's K order.
// Two contiguous 16B runs -> two b128 loads, no conversion.
// ---------------------------------------------------------------------------
__device__ __forceinline__ v16bf load_A_frag(const __bf16* __restrict__ A,
                                             int mbase, int kbase, int lane) {
  const int m  = mbase + (lane & 15);
  const int k0 = kbase + ((lane >> 4) << 4);
  const __bf16* p = A + (size_t)m * 512 + k0;
  const v8bf lo = *(const v8bf*)p;
  const v8bf hi = *(const v8bf*)(p + 8);
  v16bf f;
#pragma unroll
  for (int i = 0; i < 8; ++i) { f[i] = lo[i]; f[8 + i] = hi[i]; }
  return f;
}

// B (32x16 bf16): out = in @ W^T, so B[k][n] = W[nbase+n][kbase+k]. This code's
// contiguous K order, same as A: lane L = column n = L&15; VGPR v holds K = 2v,2v+1
// (lanes 0-15) / +16 (lanes 16-31). Element e maps to K = e + 16h, so the
// whole fragment is 16 CONTIGUOUS bf16 (32B, aligned) -> one vector load.
__device__ __forceinline__ v16bf load_B_frag(const __bf16* __restrict__ W,
                                             int nbase, int kbase, int lane) {
  const __bf16* p = W + (size_t)(nbase + (lane & 15)) * 512
                      + kbase + ((lane >> 4) << 4);
  return *(const v16bf*)p;
}

// ---------------------------------------------------------------------------
// Fused GEMM: out = act( A1 @ W1^T [+ A2 @ W2^T] + bias ), [65536x512], K=512.
// Block = 256 threads = 8 waves; each wave owns a 64(M)x64(N) strip:
// 4 M-tiles x 4 N-tiles = 16 v8f accumulators (128 VGPRs). Per K-step:
// 16 b128 loads vs 16 WMMAs (1:1 issue ratio); each B fragment feeds 4 WMMAs.
// Output: f32 (outF) or bf16 (outB) per flag.
// ---------------------------------------------------------------------------
__global__ void __launch_bounds__(256)
gemm_bf16_dual(const __bf16* __restrict__ A1, const __bf16* __restrict__ W1,
               const __bf16* __restrict__ A2, const __bf16* __restrict__ W2,
               const float* __restrict__ bias,
               float* __restrict__ outF, __bf16* __restrict__ outB, int relu) {
  const int lane = threadIdx.x & 31;
  const int wave = threadIdx.x >> 5;
  const int mblocks = NN / 512;                 // 128
  const int mb = blockIdx.x % mblocks;
  const int nb = blockIdx.x / mblocks;          // 0..7
  const int mbase = mb * 512 + wave * 64;
  const int nbase = nb * 64;

  v8f acc[4][4] = {};

  for (int kb = 0; kb < 512; kb += 32) {
    v16bf a[4];
#pragma unroll
    for (int mi = 0; mi < 4; ++mi)
      a[mi] = load_A_frag(A1, mbase + 16 * mi, kb, lane);
#pragma unroll
    for (int t = 0; t < 4; ++t) {
      const v16bf b = load_B_frag(W1, nbase + 16 * t, kb, lane);
#pragma unroll
      for (int mi = 0; mi < 4; ++mi)
        acc[mi][t] = __builtin_amdgcn_wmma_f32_16x16x32_bf16(
            false, a[mi], false, b, (short)0, acc[mi][t], false, false);
    }
  }
  if (A2 != nullptr) {
    for (int kb = 0; kb < 512; kb += 32) {
      v16bf a[4];
#pragma unroll
      for (int mi = 0; mi < 4; ++mi)
        a[mi] = load_A_frag(A2, mbase + 16 * mi, kb, lane);
#pragma unroll
      for (int t = 0; t < 4; ++t) {
        const v16bf b = load_B_frag(W2, nbase + 16 * t, kb, lane);
#pragma unroll
        for (int mi = 0; mi < 4; ++mi)
          acc[mi][t] = __builtin_amdgcn_wmma_f32_16x16x32_bf16(
              false, a[mi], false, b, (short)0, acc[mi][t], false, false);
      }
    }
  }

  // Epilogue per 32-bit C/D layout: VGPR r -> row +r+8*(lane>>4), col lane&15.
  const int rsel = (lane >> 4) << 3;
  const int ncol = lane & 15;
#pragma unroll
  for (int mi = 0; mi < 4; ++mi) {
#pragma unroll
    for (int t = 0; t < 4; ++t) {
      const int col = nbase + 16 * t + ncol;
      const float bv = bias[col];
#pragma unroll
      for (int r = 0; r < 8; ++r) {
        const int row = mbase + 16 * mi + r + rsel;
        float v = acc[mi][t][r] + bv;
        if (relu) v = fmaxf(v, 0.0f);
        if (outF) outF[(size_t)row * 512 + col] = v;
        else      outB[(size_t)row * 512 + col] = (__bf16)v;
      }
    }
  }
}

// ---------------------------------------------------------------------------
// Launch: two GraphConv+Linear+ReLU layers, bf16 WMMA with f32 accumulation.
// ws: agg_f32(128MB) | agg_bf(64MB) | x_bf/t2_bf(64MB) | t1_bf(64MB) | Wbf(3MB)
// h1_bf lives in the front half of d_out (clobbered only by the final GEMM).
// ---------------------------------------------------------------------------
extern "C" void kernel_launch(void* const* d_in, const int* in_sizes, int n_in,
                              void* d_out, int out_size, void* d_ws, size_t ws_size,
                              hipStream_t stream) {
  const float* x       = (const float*)d_in[0];
  const int*   edge    = (const int*)d_in[1];
  const int*   srcI    = edge;          // edge_index[0]
  const int*   dstI    = edge + EE;     // edge_index[1]
  const float* Wsrc[6] = {(const float*)d_in[2],  (const float*)d_in[4],
                          (const float*)d_in[5],  (const float*)d_in[7],
                          (const float*)d_in[9],  (const float*)d_in[10]};
  // order: W_rel1, W_root1, W_lin1, W_rel2, W_root2, W_lin2
  const float* b_rel1 = (const float*)d_in[3];
  const float* b_lin1 = (const float*)d_in[6];
  const float* b_rel2 = (const float*)d_in[8];
  const float* b_lin2 = (const float*)d_in[11];
  float* out = (float*)d_out;

  const size_t nelems = (size_t)NN * HH;        // 33.5M
  const size_t welems = (size_t)HH * HH;        // 262144

  float*  agg_f32 = (float*)d_ws;
  __bf16* agg_bf  = (__bf16*)(agg_f32 + nelems);
  __bf16* x_bf    = agg_bf + nelems;            // reused as t2_bf in layer 2
  __bf16* t1_bf   = x_bf + nelems;
  __bf16* Wbf     = t1_bf + nelems;             // 6 weight matrices, bf16
  __bf16* Wb[6];
  for (int i = 0; i < 6; ++i) Wb[i] = Wbf + (size_t)i * welems;
  __bf16* h1_bf = (__bf16*)d_out;               // stashed in d_out, 64MB
  __bf16* t2_bf = x_bf;

  const dim3 blk(256);
  const dim3 zgrid(2048);
  const dim3 cgrid(4096);                                        // activation cvt
  const dim3 wgrid((unsigned)(welems / 8 / 256));                // weight cvt
  const dim3 agrid((unsigned)(((size_t)EE * (FF / 8)) / 256));   // 131072
  const dim3 ggrid((NN / 512) * (HH / 64));                      // 1024

  // Pre-convert operands to bf16 (one-time, L2/HBM streaming).
  cvt_f32_bf16<<<cgrid, blk, 0, stream>>>(x, x_bf, nelems);
  for (int i = 0; i < 6; ++i)
    cvt_f32_bf16<<<wgrid, blk, 0, stream>>>(Wsrc[i], Wb[i], welems);

  // ----- layer 1 -----
  fill_zero_f32<<<zgrid, blk, 0, stream>>>(agg_f32, nelems);
  aggregate_bf16<<<agrid, blk, 0, stream>>>(x_bf, srcI, dstI, agg_f32);
  cvt_f32_bf16<<<cgrid, blk, 0, stream>>>(agg_f32, agg_bf, nelems);
  gemm_bf16_dual<<<ggrid, blk, 0, stream>>>(agg_bf, Wb[0], x_bf, Wb[1],
                                            b_rel1, nullptr, t1_bf, /*relu=*/0);
  gemm_bf16_dual<<<ggrid, blk, 0, stream>>>(t1_bf, Wb[2], nullptr, nullptr,
                                            b_lin1, nullptr, h1_bf, /*relu=*/1);

  // ----- layer 2 -----
  fill_zero_f32<<<zgrid, blk, 0, stream>>>(agg_f32, nelems);
  aggregate_bf16<<<agrid, blk, 0, stream>>>(h1_bf, srcI, dstI, agg_f32);
  cvt_f32_bf16<<<cgrid, blk, 0, stream>>>(agg_f32, agg_bf, nelems);
  gemm_bf16_dual<<<ggrid, blk, 0, stream>>>(agg_bf, Wb[3], h1_bf, Wb[4],
                                            b_rel2, nullptr, t2_bf, /*relu=*/0);
  gemm_bf16_dual<<<ggrid, blk, 0, stream>>>(t2_bf, Wb[5], nullptr, nullptr,
                                            b_lin2, out, nullptr, /*relu=*/1);
}
